// MultiHeadAttention_89824946028930
// MI455X (gfx1250) — matrix operand order repaired and checked by reading, not compiled, not run on hardware
//
#include <hip/hip_runtime.h>
#include <hip/hip_bf16.h>
#include <math.h>

// MHA forward for MI455X (gfx1250, wave32). All GEMMs via v_wmma_f32_16x16x32_bf16.
// B=4, N=4096, E=128, H=4, D=32.  Attention is flash-style (online softmax),
// K/V tiles streamed into LDS with GLOBAL_LOAD_ASYNC_TO_LDS_B128 (ASYNCcnt).

typedef __attribute__((ext_vector_type(16))) __bf16 v16bf;
typedef __attribute__((ext_vector_type(8)))  float  v8f;

#define BB   4
#define NN   4096
#define EE   128
#define HH   4
#define DD   32
#define CQKV 384
#define KBLK 64
#define KVPAD 40   // row stride in bf16: 80B = 16B-aligned, bank step 20 -> conflict-free

static __device__ __forceinline__ v8f wmma_bf16(v16bf a, v16bf b, v8f c) {
  // 8 args: (neg_a, A, neg_b, B, c_mod, C, reuse_a, reuse_b)
  return __builtin_amdgcn_wmma_f32_16x16x32_bf16(false, a, false, b, (short)0, c,
                                                 false, false);
}

// A-matrix (16x32, 16-bit) per-lane K index for vector element pair v (0..7):
// lanes 0-15: V0..3 -> K 0..7, V4..7 -> K 16..23 ; lanes 16-31: +8
static __device__ __forceinline__ int a_kidx(int v, int half) {
  return 2 * v + (v >= 4 ? 8 : 0) + half * 8;
}
// B-matrix (32x16, 16-bit): same K order as A: lanes 0-15: V0..3 -> K 0..7, V4..7 -> K 16..23 ; lanes 16-31: +8
static __device__ __forceinline__ int b_kidx(int v, int half) {
  return 2 * v + (v >= 4 ? 8 : 0) + half * 8;
}

// Async copy of 16 bytes global -> LDS (per-lane), tracked by ASYNCcnt.
static __device__ __forceinline__ void async_b128_to_lds(void* lds_ptr,
                                                         const void* gptr) {
  unsigned lds_off = (unsigned)(size_t)lds_ptr;             // addr[31:0] = LDS offset
  unsigned long long ga = (unsigned long long)(size_t)gptr; // VGPR pair
  asm volatile("global_load_async_to_lds_b128 %0, %1, off"
               :
               : "v"(lds_off), "v"(ga)
               : "memory");
}
static __device__ __forceinline__ void wait_asynccnt0() {
  asm volatile("s_wait_asynccnt 0x0" ::: "memory");
}

// ---------------------------------------------------------------------------
// Kernel 1: y = x @ W_qkv + b_qkv, de-interleave (h d qkv) -> Q/K/V[b,h,n,d] bf16.
// One 16x16 tile per wave, 4 k-steps of 32.
// ---------------------------------------------------------------------------
__global__ __launch_bounds__(128)
void qkv_wmma_kernel(const float* __restrict__ x, const float* __restrict__ Wqkv,
                     const float* __restrict__ bqkv,
                     __bf16* __restrict__ Q, __bf16* __restrict__ Kd,
                     __bf16* __restrict__ V)
{
  const int lane   = threadIdx.x & 31;
  const int wave   = threadIdx.x >> 5;
  const int tile   = blockIdx.x * 4 + wave;
  const int ntn    = CQKV / 16;              // 24 tiles along columns
  const int tile_m = tile / ntn;
  const int tile_n = tile % ntn;
  const int half   = lane >> 4;
  const int l16    = lane & 15;
  const int arow   = tile_m * 16 + l16;      // token row for A operand
  const int bcol   = tile_n * 16 + l16;      // output column for B/C

  v8f acc = {};
#pragma unroll
  for (int kk = 0; kk < EE / 32; ++kk) {
    v16bf a, bm;
#pragma unroll
    for (int v = 0; v < 8; ++v) {
      const int ka = kk * 32 + a_kidx(v, half);
      a[2 * v]     = (__bf16)x[(size_t)arow * EE + ka];
      a[2 * v + 1] = (__bf16)x[(size_t)arow * EE + ka + 1];
      const int kb = kk * 32 + b_kidx(v, half);
      bm[2 * v]     = (__bf16)Wqkv[(size_t)(kb)     * CQKV + bcol];
      bm[2 * v + 1] = (__bf16)Wqkv[(size_t)(kb + 1) * CQKV + bcol];
    }
    acc = wmma_bf16(a, bm, acc);
  }

  // Column layout is (h, d, qkv) with qkv innermost.
  const float bias  = bqkv[bcol];
  const int   h     = bcol / 96;
  const int   rem   = bcol % 96;
  const int   d     = rem / 3;
  const int   which = rem % 3;
  const float scale = 0.08838834764831845f;  // 1/sqrt(E), folded into Q
  __bf16* dst = (which == 0) ? Q : (which == 1) ? Kd : V;
  const float mul = (which == 0) ? scale : 1.0f;
#pragma unroll
  for (int j = 0; j < 8; ++j) {
    const int row = tile_m * 16 + j + 8 * half;   // global token in [0, B*N)
    const int b   = row / NN, n = row % NN;
    const float val = (acc[j] + bias) * mul;
    dst[(((size_t)b * HH + h) * NN + n) * DD + d] = (__bf16)val;
  }
}

// ---------------------------------------------------------------------------
// Kernel 2: flash attention. 8 waves/block, 16 query rows per wave,
// stream 64-key blocks through LDS via async b128 copies. S and PV via WMMA.
// ---------------------------------------------------------------------------
__global__ __launch_bounds__(256)
void attn_flash_kernel(const __bf16* __restrict__ Q, const __bf16* __restrict__ Kd,
                       const __bf16* __restrict__ V, __bf16* __restrict__ O)
{
  __shared__ __attribute__((aligned(16))) __bf16 Kt[KBLK][KVPAD]; // [key][d]
  __shared__ __attribute__((aligned(16))) __bf16 Vs[KBLK][KVPAD]; // [key][d]
  __shared__ float Sb[8][16][65];                                 // per-wave P staging

  const int lane = threadIdx.x & 31;
  const int wave = threadIdx.x >> 5;
  const int half = lane >> 4;
  const int l16  = lane & 15;

  const int nqblk = NN / 128;            // 32 query blocks per (b,h)
  const int bh    = blockIdx.x / nqblk;
  const int qblk  = blockIdx.x % nqblk;
  const int b = bh / HH, h = bh % HH;

  const size_t base = ((size_t)b * HH + h) * (size_t)NN * DD;
  const int q0 = qblk * 128 + wave * 16;

  // Load this wave's Q tile once (A operand, K = d = 32 exactly).
  v16bf qa;
#pragma unroll
  for (int v = 0; v < 8; ++v) {
    const int k = a_kidx(v, half);
    qa[2 * v]     = Q[base + (size_t)(q0 + l16) * DD + k];
    qa[2 * v + 1] = Q[base + (size_t)(q0 + l16) * DD + k + 1];
  }

  v8f o0 = {}, o1 = {};
  float m[8], lsum[8];
#pragma unroll
  for (int j = 0; j < 8; ++j) { m[j] = -1e30f; lsum[j] = 0.0f; }

  // Per-thread async-copy assignment: 8 bf16 (16B) per thread, 2048 per tile.
  const int e0   = threadIdx.x * 8;
  const int lkey = e0 / DD;
  const int ld0  = e0 % DD;              // 0, 8, 16 or 24 -> 16B-aligned LDS target

  for (int kb = 0; kb < NN; kb += KBLK) {
    // Async global -> LDS staging of K and V tiles (one b128 each per thread).
    async_b128_to_lds(&Kt[lkey][ld0], &Kd[base + (size_t)(kb + lkey) * DD + ld0]);
    async_b128_to_lds(&Vs[lkey][ld0], &V [base + (size_t)(kb + lkey) * DD + ld0]);
    if (kb + KBLK < NN) {   // prefetch next tile -> global_prefetch_b8
      __builtin_prefetch(&Kd[base + (size_t)(kb + KBLK + lkey) * DD + ld0], 0, 0);
      __builtin_prefetch(&V [base + (size_t)(kb + KBLK + lkey) * DD + ld0], 0, 0);
    }
    wait_asynccnt0();
    __syncthreads();

    // S = Qs @ K^T : four 16(q) x 16(key) tiles.
    v8f s[4];
#pragma unroll
    for (int tt = 0; tt < 4; ++tt) {
      v16bf bk;
      const int key = tt * 16 + l16;
#pragma unroll
      for (int v = 0; v < 8; ++v) {
        const int d0 = b_kidx(v, half);
        bk[2 * v]     = Kt[key][d0];
        bk[2 * v + 1] = Kt[key][d0 + 1];
      }
      v8f z = {};
      s[tt] = wmma_bf16(qa, bk, z);
    }

    // Online softmax: row stats live in 16-lane halves at fixed VGPR j.
#pragma unroll
    for (int j = 0; j < 8; ++j) {
      float tm = fmaxf(fmaxf(s[0][j], s[1][j]), fmaxf(s[2][j], s[3][j]));
#pragma unroll
      for (int mask = 1; mask < 16; mask <<= 1)
        tm = fmaxf(tm, __shfl_xor(tm, mask, 32));
      const float mn   = fmaxf(m[j], tm);
      const float corr = __expf(m[j] - mn);
      m[j] = mn;
      lsum[j] *= corr;
      o0[j] *= corr;
      o1[j] *= corr;
      float ps = 0.0f;
#pragma unroll
      for (int tt = 0; tt < 4; ++tt) {
        const float p = __expf(s[tt][j] - mn);
        Sb[wave][j + 8 * half][tt * 16 + l16] = p;   // C-layout -> LDS
        ps += p;
      }
#pragma unroll
      for (int mask = 1; mask < 16; mask <<= 1)
        ps += __shfl_xor(ps, mask, 32);
      lsum[j] += ps;
    }

    // O += P @ V : two 32-key chunks x two 16-wide d halves.
#pragma unroll
    for (int c2 = 0; c2 < 2; ++c2) {
      v16bf pa;                               // P re-read in A-layout
#pragma unroll
      for (int v = 0; v < 8; ++v) {
        const int k = c2 * 32 + a_kidx(v, half);
        pa[2 * v]     = (__bf16)Sb[wave][l16][k];
        pa[2 * v + 1] = (__bf16)Sb[wave][l16][k + 1];
      }
#pragma unroll
      for (int dh = 0; dh < 2; ++dh) {
        v16bf bv;
        const int dc = dh * 16 + l16;
#pragma unroll
        for (int v = 0; v < 8; ++v) {
          const int k = c2 * 32 + b_kidx(v, half);  // key index within tile
          bv[2 * v]     = Vs[k][dc];
          bv[2 * v + 1] = Vs[k + 1][dc];
        }
        if (dh == 0) o0 = wmma_bf16(pa, bv, o0);
        else         o1 = wmma_bf16(pa, bv, o1);
      }
    }
    __syncthreads();   // protect Kt/Vs before next block's async loads
  }

  // Normalize and stage O as bf16 rows [b*N + n][h*D + d] for the projection.
#pragma unroll
  for (int j = 0; j < 8; ++j) {
    const float inv = 1.0f / lsum[j];
    const int n = q0 + j + 8 * half;
    const size_t rowo = ((size_t)b * NN + n) * EE + (size_t)h * DD;
    O[rowo + l16]      = (__bf16)(o0[j] * inv);
    O[rowo + 16 + l16] = (__bf16)(o1[j] * inv);
  }
}

// ---------------------------------------------------------------------------
// Kernel 3: out = O @ W_proj + b_proj (f32 output).
// ---------------------------------------------------------------------------
__global__ __launch_bounds__(128)
void proj_wmma_kernel(const __bf16* __restrict__ O, const float* __restrict__ Wp,
                      const float* __restrict__ bp, float* __restrict__ out)
{
  const int lane   = threadIdx.x & 31;
  const int wave   = threadIdx.x >> 5;
  const int tile   = blockIdx.x * 4 + wave;
  const int ntn    = EE / 16;                // 8
  const int tile_m = tile / ntn;
  const int tile_n = tile % ntn;
  const int half   = lane >> 4;
  const int l16    = lane & 15;
  const int arow   = tile_m * 16 + l16;
  const int bcol   = tile_n * 16 + l16;

  v8f acc = {};
#pragma unroll
  for (int kk = 0; kk < EE / 32; ++kk) {
    v16bf a, bm;
#pragma unroll
    for (int v = 0; v < 8; ++v) {
      const int ka = kk * 32 + a_kidx(v, half);
      a[2 * v]     = O[(size_t)arow * EE + ka];
      a[2 * v + 1] = O[(size_t)arow * EE + ka + 1];
      const int kb = kk * 32 + b_kidx(v, half);
      bm[2 * v]     = (__bf16)Wp[(size_t)(kb)     * EE + bcol];
      bm[2 * v + 1] = (__bf16)Wp[(size_t)(kb + 1) * EE + bcol];
    }
    acc = wmma_bf16(a, bm, acc);
  }
  const float bias = bp[bcol];
#pragma unroll
  for (int j = 0; j < 8; ++j) {
    const int row = tile_m * 16 + j + 8 * half;
    out[(size_t)row * EE + bcol] = acc[j] + bias;
  }
}

// ---------------------------------------------------------------------------
extern "C" void kernel_launch(void* const* d_in, const int* in_sizes, int n_in,
                              void* d_out, int out_size, void* d_ws, size_t ws_size,
                              hipStream_t stream) {
  const float* x     = (const float*)d_in[0];
  const float* W_qkv = (const float*)d_in[1];
  const float* b_qkv = (const float*)d_in[2];
  const float* W_prj = (const float*)d_in[3];
  const float* b_prj = (const float*)d_in[4];
  float* out = (float*)d_out;

  const size_t hd = (size_t)BB * HH * NN * DD;   // 2,097,152 elems per tensor
  __bf16* Q = (__bf16*)d_ws;
  __bf16* K = Q + hd;
  __bf16* V = K + hd;
  __bf16* O = V + hd;                            // [B*N, E] bf16

  // QKV projection: (B*N/16)*(384/16) = 24576 wave-tiles, 4 waves/block.
  qkv_wmma_kernel<<<(BB * NN / 16) * (CQKV / 16) / 4, 128, 0, stream>>>(
      x, W_qkv, b_qkv, Q, K, V);

  // Flash attention: B*H*(N/128) = 512 blocks of 8 waves.
  attn_flash_kernel<<<BB * HH * (NN / 128), 256, 0, stream>>>(Q, K, V, O);

  // Output projection: (B*N/16)*(128/16) = 8192 wave-tiles, 4 waves/block.
  proj_wmma_kernel<<<(BB * NN / 16) * (EE / 16) / 4, 128, 0, stream>>>(
      O, W_prj, b_prj, out);
}
